// SOMISettlingLayer_33200097198709
// MI455X (gfx1250) — compile-verified, matrix operand order checked by reading, not run on hardware
//
#include <hip/hip_runtime.h>
#include <hip/hip_bf16.h>
#include <math.h>

// ---------------- SOMI constants (match reference) ----------------
#define SOMI_D      512
#define SOMI_ROWS   16384          // B*S = 4*4096
#define SOMI_MTILE  32             // rows per workgroup
#define SOMI_STEPS  10             // N_STEPS from reference
#define ALPHA0_F    0.1f
#define DT_F        0.1f
#define BETA_F      2.8982753492378876f   // 2*sqrt(2.1)

typedef __attribute__((ext_vector_type(8)))  __bf16 bf16x8_t;
typedef __attribute__((ext_vector_type(16))) __bf16 bf16x16_t;
typedef __attribute__((ext_vector_type(8)))  float  f32x8_t;

// Load one 16x32 bf16 WMMA operand fragment for this lane.
// Per CDNA5 ISA layout: lane&15 selects row (A) / col (B); lane>=16 shifts the
// K-window by +8; the two 8-element halves are K+0..7 and K+16..23.
__device__ __forceinline__ bf16x16_t ldfrag(const __bf16* p) {
    union { bf16x16_t v; bf16x8_t h[2]; } u;
    u.h[0] = *(const bf16x8_t*)(p);
    u.h[1] = *(const bf16x8_t*)(p + 16);
    return u.v;
}

// ---------------- Kernel 1: build bf16 random-walk Laplacian + gate ---------
// Lb[n][k] = (n==k) - |W[n][k]| / max(sum_k |W[n][k]|, 1e-8), row-major bf16.
// Note: WMMA B operand is L_rw^T, and a B-fragment lane reads a fixed output
// column n over contiguous K — which is exactly row n of Lb. No transpose needed.
__global__ void somi_prep_kernel(const float* __restrict__ W,
                                 const float* __restrict__ gate_alpha,
                                 float* __restrict__ gate_out,
                                 __bf16* __restrict__ Lb) {
    const int n   = blockIdx.x;          // 512 blocks, one per row of W
    const int tid = threadIdx.x;         // 256 threads
    __shared__ float red[256];

    float s = 0.f;
    for (int k = tid; k < SOMI_D; k += 256)
        s += fabsf(W[n * SOMI_D + k]);
    red[tid] = s;
    __syncthreads();
    for (int off = 128; off > 0; off >>= 1) {
        if (tid < off) red[tid] += red[tid + off];
        __syncthreads();
    }
    const float inv = 1.0f / fmaxf(red[0], 1e-8f);

    for (int k = tid; k < SOMI_D; k += 256) {
        float v = ((k == n) ? 1.0f : 0.0f) - fabsf(W[n * SOMI_D + k]) * inv;
        Lb[n * SOMI_D + k] = (__bf16)v;
    }
    if (n == 0 && tid == 0) *gate_out = tanhf(gate_alpha[0]);
}

// ---------------- Kernel 2: fused 10-step settle, WMMA bf16 -----------------
// Grid: 512 workgroups x 1024 threads (32 waves). Each WG owns 32 rows.
// Wave w: mtile = w>>4 (rows 16*mtile..+15), ntiles = 2*(w&15) + {0,1}.
// Each lane's 8 accumulator regs per tile map to (m = mtile*16 + 8*hi + r,
// n = ntile*16 + (lane&15)) — the pointwise Euler state is kept in registers
// in exactly this layout, so only the bf16 LDS mirror of phi is shared.
__global__ __launch_bounds__(1024, 1)
void somi_settle_kernel(const float* __restrict__ hidden,
                        const float* __restrict__ precision,
                        const __bf16* __restrict__ Lb,
                        const float* __restrict__ gate_p,
                        float* __restrict__ out) {
    __shared__ __align__(16) __bf16 sPhi[SOMI_MTILE * SOMI_D];

    const int tid    = threadIdx.x;
    const int wave   = tid >> 5;
    const int lane   = tid & 31;
    const int lane15 = lane & 15;
    const int hi     = lane >> 4;            // 0 or 1
    const int mtile  = wave >> 4;            // 0 or 1
    const int ntb    = (wave & 15) * 2;      // base n-tile index (each 16 wide)
    const int mloc0  = mtile * 16 + hi * 8;  // local row of accumulator reg r=0
    const long rowBase = (long)blockIdx.x * SOMI_MTILE;

    float phi[2][8], pdot[2][8], targ[2][8], prec[2];

    // ---- initial state load (phi0 = target = hidden), seed bf16 LDS mirror ----
    #pragma unroll
    for (int t = 0; t < 2; ++t) {
        const int n = (ntb + t) * 16 + lane15;
        prec[t] = precision[n];
        #pragma unroll
        for (int r = 0; r < 8; ++r) {
            const int m = mloc0 + r;
            const float v = hidden[(rowBase + m) * SOMI_D + n];
            targ[t][r] = v; phi[t][r] = v; pdot[t][r] = 0.f;
            sPhi[m * SOMI_D + n] = (__bf16)v;
        }
    }
    const float gate = *gate_p;

    // Fragment base pointers (lane-resolved once).
    const __bf16* aBase  = sPhi + (mtile * 16 + lane15) * SOMI_D + hi * 8;
    const __bf16* b0Base = Lb + (long)((ntb + 0) * 16 + lane15) * SOMI_D + hi * 8;
    const __bf16* b1Base = Lb + (long)((ntb + 1) * 16 + lane15) * SOMI_D + hi * 8;
    __builtin_prefetch(b0Base, 0, 3);   // warm L2 for the L2-resident Laplacian
    __builtin_prefetch(b1Base, 0, 3);

    __syncthreads();

    // ---- 10 fused semi-implicit Euler settling steps ----
    for (int step = 0; step < SOMI_STEPS; ++step) {
        f32x8_t c0 = {}; f32x8_t c1 = {};
        // coupling = phi @ L_rw^T : K = 512 = 16 WMMA slabs of 32
        #pragma unroll 4
        for (int k = 0; k < 16; ++k) {
            const bf16x16_t a  = ldfrag(aBase  + k * 32);
            const bf16x16_t b0 = ldfrag(b0Base + k * 32);
            const bf16x16_t b1 = ldfrag(b1Base + k * 32);
            c0 = __builtin_amdgcn_wmma_f32_16x16x32_bf16(false, a, false, b0,
                                                         (short)0, c0, false, false);
            c1 = __builtin_amdgcn_wmma_f32_16x16x32_bf16(false, a, false, b1,
                                                         (short)0, c1, false, false);
        }
        __syncthreads();   // all waves done reading sPhi for this step

        // pointwise: force -> phi_dot -> phi, fully register-local
        #pragma unroll
        for (int t = 0; t < 2; ++t) {
            #pragma unroll
            for (int r = 0; r < 8; ++r) {
                const float cp = (t == 0) ? c0[r] : c1[r];
                const float f  = -cp - ALPHA0_F * phi[t][r]
                                 + prec[t] * (targ[t][r] - phi[t][r])
                                 - BETA_F * pdot[t][r];
                pdot[t][r] += DT_F * f;
                phi[t][r]  += DT_F * pdot[t][r];
            }
        }
        // refresh bf16 LDS mirror of phi
        #pragma unroll
        for (int t = 0; t < 2; ++t) {
            const int n = (ntb + t) * 16 + lane15;
            #pragma unroll
            for (int r = 0; r < 8; ++r)
                sPhi[(mloc0 + r) * SOMI_D + n] = (__bf16)phi[t][r];
        }
        __syncthreads();   // mirror visible before next step's A-fragment reads
    }

    // ---- out = hidden + gate * (phi_settled - hidden) ----
    #pragma unroll
    for (int t = 0; t < 2; ++t) {
        const int n = (ntb + t) * 16 + lane15;
        #pragma unroll
        for (int r = 0; r < 8; ++r) {
            const int m = mloc0 + r;
            out[(rowBase + m) * SOMI_D + n] =
                targ[t][r] + gate * (phi[t][r] - targ[t][r]);
        }
    }
}

// ---------------- launcher ----------------
extern "C" void kernel_launch(void* const* d_in, const int* in_sizes, int n_in,
                              void* d_out, int out_size, void* d_ws, size_t ws_size,
                              hipStream_t stream) {
    const float* hidden     = (const float*)d_in[0];   // (4,4096,512) f32
    const float* W          = (const float*)d_in[1];   // (512,512) f32
    const float* gate_alpha = (const float*)d_in[2];   // (1,) f32
    const float* precision  = (const float*)d_in[3];   // (512,) f32
    float* out = (float*)d_out;

    // workspace: [0..3] gate scalar, [256..] bf16 Laplacian (512 KB)
    float*  gate_ws = (float*)d_ws;
    __bf16* Lb      = (__bf16*)((char*)d_ws + 256);

    somi_prep_kernel<<<SOMI_D, 256, 0, stream>>>(W, gate_alpha, gate_ws, Lb);
    somi_settle_kernel<<<SOMI_ROWS / SOMI_MTILE, 1024, 0, stream>>>(
        hidden, precision, Lb, gate_ws, out);
}
